// Block_3977139716684
// MI455X (gfx1250) — matrix operand order repaired and checked by reading, not compiled, not run on hardware
//
#include <hip/hip_runtime.h>
#include <stdint.h>

#define Bn   8
#define Cc   192
#define OUTC 384
#define Nn   3136          // 56*56
#define Kk   16
#define C2   384           // 2*Cc
#define EPSV 1e-5f

typedef __attribute__((ext_vector_type(16))) __bf16       v16bf;
typedef __attribute__((ext_vector_type(8)))  float        v8f;
typedef __attribute__((ext_vector_type(8)))  unsigned int v8u;

static __device__ __forceinline__ unsigned short f2bf(float f) {
  unsigned int u = __float_as_uint(f);
  u += 0x7fffu + ((u >> 16) & 1u);          // round-to-nearest-even
  return (unsigned short)(u >> 16);
}

// ---------------------------------------------------------------- prep ----
// x (B,C,N) -> xf f32 (B,N,C) and bf16 copy, via 32x33 LDS tile transpose
__global__ void __launch_bounds__(256)
transpose_kernel(const float* __restrict__ x, float* __restrict__ xfF,
                 unsigned short* __restrict__ xf16) {
  __shared__ float sm[32][33];
  int blk = blockIdx.x;
  int tN = blk % 98; blk /= 98;             // Nn/32 = 98
  int tC = blk % 6;  blk /= 6;              // Cc/32 = 6
  int bb = blk;
  int tx = threadIdx.x & 31, ty = threadIdx.x >> 5;
  int c0 = tC * 32, n0 = tN * 32;
#pragma unroll
  for (int r = 0; r < 4; ++r) {
    int c = c0 + ty + r * 8;
    sm[ty + r * 8][tx] = x[((size_t)bb * Cc + c) * Nn + n0 + tx];
  }
  __syncthreads();
#pragma unroll
  for (int r = 0; r < 4; ++r) {
    int n = n0 + ty + r * 8;
    float v = sm[tx][ty + r * 8];
    size_t o = ((size_t)bb * Nn + n) * Cc + c0 + tx;
    xfF[o]  = v;
    xf16[o] = f2bf(v);
  }
}

__global__ void __launch_bounds__(256)
x2_kernel(const float* __restrict__ x, float* __restrict__ x2) {
  int bb = blockIdx.x / 13;
  int n  = (blockIdx.x % 13) * 256 + threadIdx.x;
  if (n >= Nn) return;
  float acc = 0.f;
  for (int c = 0; c < Cc; ++c) {
    float v = x[((size_t)bb * Cc + c) * Nn + n];
    acc += v * v;
  }
  x2[bb * Nn + n] = acc;
}

__global__ void __launch_bounds__(256)
cvtw_kernel(const float* __restrict__ w, unsigned short* __restrict__ w16) {
  int i = blockIdx.x * 256 + threadIdx.x;
  if (i < OUTC * C2) w16[i] = f2bf(w[i]);
}

// ------------------------------------------------------------ fused kNN ----
// block = 256 thr (8 waves); each wave owns a 16-row strip of a 128-row tile.
// Streams 16-column tiles: 6x v_wmma_f32_16x16x32_bf16 per tile, then per-row
// sorted top-16 (registers) on key = x2_j - 2*inner (x2_i is rank-invariant).
#define BROWS 128
__global__ void __launch_bounds__(256)
knn_kernel(const unsigned short* __restrict__ xf16, const float* __restrict__ x2,
           int* __restrict__ idxOut) {
  __shared__ __align__(16) unsigned short sRows[BROWS * Cc];  // 48 KB
  __shared__ __align__(16) unsigned short sCols[16 * Cc];     // 6 KB
  __shared__ float sX2[16];
  __shared__ __align__(16) float sKeys[8 * 256];              // per-wave 16x16
  int bb = blockIdx.y;
  int row0 = blockIdx.x * BROWS;
  int tid = threadIdx.x;
  int lane = tid & 31, w = tid >> 5;
  int ln = lane & 15, hi = lane >> 4;

  { // stage 128 rows (b128 loads), zero-fill OOB rows
    const uint4* g = (const uint4*)(xf16 + ((size_t)bb * Nn + row0) * Cc);
    uint4* s = (uint4*)sRows;
    for (int i = tid; i < BROWS * Cc / 8; i += 256) {
      int r = i / 24;                        // Cc/8 = 24 uint4 per row
      uint4 v = make_uint4(0u, 0u, 0u, 0u);
      if (row0 + r < Nn) v = g[i];
      s[i] = v;
    }
  }
  __syncthreads();

  int m0 = w * 16;
  bool valid = (row0 + m0) < Nn;             // wave-uniform

  // resident A fragments: lane(0-15)=row, hi selects K 0-7/16-23 vs 8-15/24-31
  v16bf afrag[6];
  if (valid) {
    int lr = m0 + ln;
#pragma unroll
    for (int kk = 0; kk < 6; ++kk) {
      int base = lr * Cc + kk * 32 + hi * 8;
      uint4 q0 = *(const uint4*)&sRows[base];
      uint4 q1 = *(const uint4*)&sRows[base + 16];
      v8u u = {q0.x, q0.y, q0.z, q0.w, q1.x, q1.y, q1.z, q1.w};
      afrag[kk] = __builtin_bit_cast(v16bf, u);
    }
  }

  float tk[16]; int ti[16];
#pragma unroll
  for (int p = 0; p < 16; ++p) { tk[p] = 3.0e38f; ti[p] = 0; }

  for (int j0 = 0; j0 < Nn; j0 += 16) {
    { // stage 16 columns + their norms
      const uint4* g = (const uint4*)(xf16 + ((size_t)bb * Nn + j0) * Cc);
      uint4* s = (uint4*)sCols;
      for (int i = tid; i < 16 * Cc / 8; i += 256) s[i] = g[i];
      if (tid < 16) sX2[tid] = x2[bb * Nn + j0 + tid];
    }
    __syncthreads();

    if (valid) {
      v8f acc = {0.f, 0.f, 0.f, 0.f, 0.f, 0.f, 0.f, 0.f};
#pragma unroll
      for (int kk = 0; kk < 6; ++kk) {
        int base = ln * Cc + kk * 32 + hi * 8;    // B: lane=col, hi selects K 0-7/16-23 vs 8-15/24-31
        uint4 q0 = *(const uint4*)&sCols[base];
        uint4 q1 = *(const uint4*)&sCols[base + 16];
        v8u u = {q0.x, q0.y, q0.z, q0.w, q1.x, q1.y, q1.z, q1.w};
        v16bf bfrag = __builtin_bit_cast(v16bf, u);
        acc = __builtin_amdgcn_wmma_f32_16x16x32_bf16(
            false, afrag[kk], false, bfrag, (short)0, acc, false, false);
      }
      float x2c = sX2[ln];
#pragma unroll
      for (int r = 0; r < 8; ++r)                 // D: row = r + hi*8, col = ln
        sKeys[w * 256 + (r + hi * 8) * 16 + ln] = fmaf(-2.0f, acc[r], x2c);

      if (lane < 16) {                            // lane owns one row
        // pull the whole row back with 4x ds_load_b128 (one wait), scan in regs
        const float4* kp = (const float4*)&sKeys[w * 256 + lane * 16];
        float kv[16];
#pragma unroll
        for (int q = 0; q < 4; ++q) {
          float4 t = kp[q];
          kv[4 * q + 0] = t.x; kv[4 * q + 1] = t.y;
          kv[4 * q + 2] = t.z; kv[4 * q + 3] = t.w;
        }
#pragma unroll
        for (int c16 = 0; c16 < 16; ++c16) {
          float v = kv[c16];
          int j = j0 + c16;
          if (v < tk[15]) {                       // replace worst + bubble
            tk[15] = v; ti[15] = j;
#pragma unroll
            for (int p = 15; p > 0; --p) {
              if (tk[p] < tk[p - 1]) {
                float tf = tk[p]; tk[p] = tk[p - 1]; tk[p - 1] = tf;
                int   tj = ti[p]; ti[p] = ti[p - 1]; ti[p - 1] = tj;
              }
            }
          }
        }
      }
    }
    __syncthreads();
  }

  if (valid && lane < 16) {
    int row = row0 + m0 + lane;
#pragma unroll
    for (int p = 0; p < 16; ++p)
      idxOut[((size_t)bb * Nn + row) * Kk + p] = ti[p];
  }
}

// -------------------------------------------------- max-relative + feat ----
__global__ void __launch_bounds__(192)
mrconv_kernel(const float* __restrict__ xfF, const int* __restrict__ idx,
              unsigned int* __restrict__ feat16) {
  __shared__ int sIdx[16];
  int bb = blockIdx.x / Nn;
  int n  = blockIdx.x % Nn;
  if (threadIdx.x < 16) sIdx[threadIdx.x] = idx[((size_t)bb * Nn + n) * Kk + threadIdx.x];
  __syncthreads();
  int c = threadIdx.x;
  float xn = xfF[((size_t)bb * Nn + n) * Cc + c];
  float m = -3.0e38f;
#pragma unroll 4
  for (int k = 0; k < Kk; ++k) {
    int j = sIdx[k];
    m = fmaxf(m, xfF[((size_t)bb * Nn + j) * Cc + c] - xn);
  }
  // interleaved (c_x, c_rel) bf16 pair packed in one dword
  unsigned int packed = (unsigned int)f2bf(xn) | ((unsigned int)f2bf(m) << 16);
  feat16[((size_t)bb * Nn + n) * Cc + c] = packed;
}

// ------------------------------------------------------- bf16 WMMA GEMM ----
// y = feat(25088x384) @ w^T(384x384) + b ; 64-row tiles, 12 k-steps.
#define DROWS 64
__global__ void __launch_bounds__(256)
gemm_kernel(const unsigned short* __restrict__ feat16,
            const unsigned short* __restrict__ w16,
            const float* __restrict__ bias, float* __restrict__ yPre) {
  __shared__ __align__(16) unsigned short sFeat[DROWS * C2];   // 48 KB
  int bb = blockIdx.y;
  int row0 = blockIdx.x * DROWS;                               // 49*64 = 3136
  int tid = threadIdx.x;
  int lane = tid & 31, w = tid >> 5;
  int ln = lane & 15, hi = lane >> 4;
  {
    const uint4* g = (const uint4*)(feat16 + ((size_t)bb * Nn + row0) * C2);
    uint4* s = (uint4*)sFeat;
    for (int i = tid; i < DROWS * C2 / 8; i += 256) s[i] = g[i];
  }
  __syncthreads();
  int strip = w & 3, half = w >> 2;          // 2 waves share a 16-row strip
  int m0 = strip * 16;
  int lr = m0 + ln;
  for (int t = 0; t < 12; ++t) {
    int o0 = (half * 12 + t) * 16;
    int o  = o0 + ln;
    v8f acc = {0.f, 0.f, 0.f, 0.f, 0.f, 0.f, 0.f, 0.f};
#pragma unroll
    for (int kk = 0; kk < 12; ++kk) {
      int ab = lr * C2 + kk * 32 + hi * 8;
      uint4 a0 = *(const uint4*)&sFeat[ab];
      uint4 a1 = *(const uint4*)&sFeat[ab + 16];
      v8u au = {a0.x, a0.y, a0.z, a0.w, a1.x, a1.y, a1.z, a1.w};
      v16bf af = __builtin_bit_cast(v16bf, au);
      int wb = o * C2 + kk * 32 + hi * 8;    // w row-major (o, c2): B = w^T
      uint4 b0 = *(const uint4*)&w16[wb];
      uint4 b1 = *(const uint4*)&w16[wb + 16];
      v8u bu = {b0.x, b0.y, b0.z, b0.w, b1.x, b1.y, b1.z, b1.w};
      v16bf bf = __builtin_bit_cast(v16bf, bu);
      acc = __builtin_amdgcn_wmma_f32_16x16x32_bf16(
          false, af, false, bf, (short)0, acc, false, false);
    }
    float bv = bias[o];
#pragma unroll
    for (int r = 0; r < 8; ++r) {
      int row = row0 + m0 + r + hi * 8;
      yPre[((size_t)bb * Nn + row) * OUTC + o] = acc[r] + bv;
    }
  }
}

// ----------------------------------------------- BN stats (deterministic) ----
__global__ void __launch_bounds__(256)
stats_kernel(const float* __restrict__ yPre, float* __restrict__ stats) {
  __shared__ float sS[256], sQ[256];
  int ch = blockIdx.x;
  float s = 0.f, q = 0.f;
  for (int i = threadIdx.x; i < Bn * Nn; i += 256) {
    float v = yPre[(size_t)i * OUTC + ch];
    s += v; q += v * v;
  }
  sS[threadIdx.x] = s; sQ[threadIdx.x] = q;
  __syncthreads();
  for (int st = 128; st > 0; st >>= 1) {
    if (threadIdx.x < st) {
      sS[threadIdx.x] += sS[threadIdx.x + st];
      sQ[threadIdx.x] += sQ[threadIdx.x + st];
    }
    __syncthreads();
  }
  if (threadIdx.x == 0) { stats[ch] = sS[0]; stats[OUTC + ch] = sQ[0]; }
}

// ------------------------------------------- BN apply + ReLU + transpose ----
__global__ void __launch_bounds__(256)
bn_kernel(const float* __restrict__ yPre, const float* __restrict__ stats,
          const float* __restrict__ gamma, const float* __restrict__ beta,
          float* __restrict__ out) {
  int bb = blockIdx.x / OUTC;
  int o  = blockIdx.x % OUTC;
  const float inv = 1.0f / (float)(Bn * Nn);
  float mean = stats[o] * inv;
  float var  = stats[OUTC + o] * inv - mean * mean;   // biased, torch-style
  float sc = gamma[o] * rsqrtf(var + EPSV);
  float sh = beta[o] - mean * sc;
  for (int n = threadIdx.x; n < Nn; n += 256) {
    float v = yPre[((size_t)bb * Nn + n) * OUTC + o];
    out[((size_t)bb * OUTC + o) * Nn + n] = fmaxf(v * sc + sh, 0.f);
  }
}

// ------------------------------------------------------------- launcher ----
extern "C" void kernel_launch(void* const* d_in, const int* in_sizes, int n_in,
                              void* d_out, int out_size, void* d_ws, size_t ws_size,
                              hipStream_t stream) {
  const float* x     = (const float*)d_in[0];
  const float* w     = (const float*)d_in[1];
  const float* b     = (const float*)d_in[2];
  const float* gamma = (const float*)d_in[3];
  const float* beta  = (const float*)d_in[4];
  float* out = (float*)d_out;

  char* ws = (char*)d_ws;
  float* xfF = (float*)ws;                    ws += (size_t)Bn * Nn * Cc * 4;
  float* x2  = (float*)ws;                    ws += (size_t)Bn * Nn * 4;
  unsigned short* xf16 = (unsigned short*)ws; ws += (size_t)Bn * Nn * Cc * 2;
  unsigned short* w16  = (unsigned short*)ws; ws += (size_t)OUTC * C2 * 2;
  int* idx = (int*)ws;                        ws += (size_t)Bn * Nn * Kk * 4;
  unsigned short* feat16 = (unsigned short*)ws; ws += (size_t)Bn * Nn * C2 * 2;
  float* yPre  = (float*)ws;                  ws += (size_t)Bn * Nn * OUTC * 4;
  float* stats = (float*)ws;

  transpose_kernel<<<dim3(Bn * 6 * 98), dim3(256), 0, stream>>>(x, xfF, xf16);
  x2_kernel<<<dim3(Bn * 13), dim3(256), 0, stream>>>(x, x2);
  cvtw_kernel<<<dim3((OUTC * C2 + 255) / 256), dim3(256), 0, stream>>>(w, w16);
  knn_kernel<<<dim3(25, Bn), dim3(256), 0, stream>>>(xf16, x2, idx);
  mrconv_kernel<<<dim3(Bn * Nn), dim3(192), 0, stream>>>(xfF, idx, (unsigned int*)feat16);
  gemm_kernel<<<dim3(49, Bn), dim3(256), 0, stream>>>(feat16, w16, b, yPre);
  stats_kernel<<<dim3(OUTC), dim3(256), 0, stream>>>(yPre, stats);
  bn_kernel<<<dim3(Bn * OUTC), dim3(256), 0, stream>>>(yPre, stats, gamma, beta, out);
}
